// SoftMaxHungarianMatcher_34196529610818
// MI455X (gfx1250) — matrix operand order repaired and checked by reading, not compiled, not run on hardware
//
#include <hip/hip_runtime.h>
#include <hip/hip_bf16.h>

// ---------------- Problem constants ----------------
#define TT 36
#define QQ 2048
#define NN 128
#define CC 42
#define KDIM (TT * CC)      // 1512
#define KPAD 1536           // padded K (multiple of 32)
#define NUMCLS (CC - 1)     // 41

typedef __attribute__((ext_vector_type(16))) __bf16 v16bf;
typedef __attribute__((ext_vector_type(8)))  float  v8f;

union ABFrag { v16bf v; unsigned int u[8]; };

__device__ __forceinline__ unsigned short f2bf(float f) {
    unsigned int u = __float_as_uint(f);
    u += 0x7FFFu + ((u >> 16) & 1u);   // round-to-nearest-even
    return (unsigned short)(u >> 16);
}

// ---------------- Kernel 1: zero Bt and A pad columns ----------------
__global__ __launch_bounds__(256) void init_ws_kernel(unsigned short* A, unsigned short* Bt) {
    int idx = blockIdx.x * 256 + threadIdx.x;
    if (idx < NN * KPAD) Bt[idx] = 0;                       // whole Bt (n,k) zeroed
    if (idx < QQ * (KPAD - KDIM)) {                         // A pad cols 1512..1535
        int q = idx / (KPAD - KDIM);
        int c = idx % (KPAD - KDIM);
        A[q * KPAD + KDIM + c] = 0;
    }
}

// ---------------- Kernel 2: softmax rows -> bf16 A[q, t*C+c] ----------------
__global__ __launch_bounds__(256) void softmax_kernel(const float* __restrict__ logits,
                                                      unsigned short* __restrict__ A) {
    int row  = blockIdx.x * 8 + (threadIdx.x >> 5);   // row = t*Q + q, 73728 rows
    int lane = threadIdx.x & 31;
    const float* src = logits + row * CC;

    float x0 = src[lane];
    float x1 = (lane < CC - 32) ? src[lane + 32] : -INFINITY;

    float m = fmaxf(x0, x1);
    #pragma unroll
    for (int off = 16; off > 0; off >>= 1) m = fmaxf(m, __shfl_xor(m, off, 32));

    float e0 = __expf(x0 - m);
    float e1 = (lane < CC - 32) ? __expf(x1 - m) : 0.0f;
    float s = e0 + e1;
    #pragma unroll
    for (int off = 16; off > 0; off >>= 1) s += __shfl_xor(s, off, 32);
    float inv = 1.0f / s;

    int t = row / QQ;
    int q = row - t * QQ;
    unsigned short* dst = A + q * KPAD + t * CC;
    dst[lane] = f2bf(e0 * inv);
    if (lane < CC - 32) dst[lane + 32] = f2bf(e1 * inv);
}

// ---------------- Kernel 3: scatter one-hot weights into Bt[n, k] ----------------
__global__ __launch_bounds__(256) void scatter_b_kernel(const int* __restrict__ labels,
                                                        const int* __restrict__ valid,
                                                        const int* __restrict__ ovalid,
                                                        unsigned short* __restrict__ Bt) {
    int idx = blockIdx.x * 256 + threadIdx.x;   // idx = n*T + t
    if (idx >= NN * TT) return;
    int n = idx / TT;
    int t = idx - n * TT;

    float denom = 0.0f;
    #pragma unroll 4
    for (int tt = 0; tt < TT; ++tt) denom += (float)valid[n * TT + tt];

    int id = (ovalid[idx] == 0) ? NUMCLS : labels[idx];
    int v  = valid[idx];
    float w = (v != 0 && denom > 0.0f) ? (-(float)v / denom) : 0.0f;
    Bt[n * KPAD + t * CC + id] = f2bf(w);
}

// ---------------- Kernel 4: bbox L1 + IoU costs -> d_out ----------------
__global__ __launch_bounds__(256) void boxcost_kernel(const float* __restrict__ pred_boxes,
                                                      const float* __restrict__ tgt_boxes,
                                                      const int* __restrict__ valid,
                                                      float* __restrict__ out) {
    int idx = blockIdx.x * 256 + threadIdx.x;   // idx = q*N + n
    int q = idx >> 7;
    int n = idx & (NN - 1);

    const float4* pb = (const float4*)pred_boxes;
    const float4* tb = (const float4*)tgt_boxes;

    float sb = 0.0f, si = 0.0f, dv = 0.0f;
    #pragma unroll 4
    for (int t = 0; t < TT; ++t) {
        float v = (float)valid[n * TT + t];
        float4 b = pb[t * QQ + q];      // boxes[q,t] cxcywh
        float4 g = tb[n * TT + t];      // tboxes[n,t] cxcywh

        // L1 on cxcywh (mean of 4 coords)
        float l1 = 0.25f * (fabsf(b.x - g.x) + fabsf(b.y - g.y) +
                            fabsf(b.z - g.z) + fabsf(b.w - g.w));

        // convert to xyxy
        float bx0 = b.x - 0.5f * b.z, by0 = b.y - 0.5f * b.w;
        float bx1 = b.x + 0.5f * b.z, by1 = b.y + 0.5f * b.w;
        float gx0 = g.x - 0.5f * g.z, gy0 = g.y - 0.5f * g.w;
        float gx1 = g.x + 0.5f * g.z, gy1 = g.y + 0.5f * g.w;

        float ltx = fmaxf(bx0, gx0), lty = fmaxf(by0, gy0);
        float rbx = fminf(bx1, gx1), rby = fminf(by1, gy1);
        float iw = fmaxf(rbx - ltx, 0.0f), ih = fmaxf(rby - lty, 0.0f);
        float inter = iw * ih;
        float a1 = (bx1 - bx0) * (by1 - by0);
        float a2 = (gx1 - gx0) * (gy1 - gy0);
        float uni = a1 + a2 - inter;
        float iou = inter / uni;

        sb += v * l1;
        si += v * iou;
        dv += v;
    }
    out[idx] = (sb - si) / dv;   // bbox_cost + iou_cost (class cost added by GEMM)
}

// ---------------- Kernel 5: WMMA GEMM, out += A(QxK) * B(KxN) ----------------
// One wave per 16x16 output tile. A is (q, k) bf16 row-major; B stored transposed
// Bt (n, k) so both fragments load as contiguous bf16 pairs (dword loads).
__global__ __launch_bounds__(256) void class_gemm_kernel(const unsigned short* __restrict__ A,
                                                         const unsigned short* __restrict__ Bt,
                                                         float* __restrict__ out) {
    int wave = threadIdx.x >> 5;
    int lane = threadIdx.x & 31;
    int tile = blockIdx.x * 8 + wave;        // 1024 tiles total
    int mt = tile >> 3;                      // 128 M-tiles
    int nt = tile & 7;                       // 8 N-tiles
    int q0 = mt * 16;
    int n0 = nt * 16;

    int half  = lane >> 4;                   // lane group 0/1
    int idx16 = lane & 15;

    const unsigned short* arow = A  + (q0 + idx16) * KPAD;  // A: M = idx16
    const unsigned short* brow = Bt + (n0 + idx16) * KPAD;  // B: N = idx16

    // C fragment from boxcost result: lane idx16 = N, vgpr r -> M = r + 8*half
    float* obase = out + q0 * NN + n0 + idx16;
    v8f c;
    #pragma unroll
    for (int r = 0; r < 8; ++r) c[r] = obase[(r + 8 * half) * NN];

    for (int k0 = 0; k0 < KPAD; k0 += 32) {
        ABFrag a, b;
        #pragma unroll
        for (int j = 0; j < 8; ++j) {
            // A 16-bit 16x32 layout: lanes0-15 K={0..7,16..23}, lanes16-31 +8
            int ka = k0 + ((j < 4) ? (2 * j) : (2 * j + 8)) + 8 * half;
            a.u[j] = *(const unsigned int*)(arow + ka);
            // B 16-bit 32x16 layout: lanes0-15 K={0..7,16..23}, lanes16-31 +8
            int kb = k0 + ((j < 4) ? (2 * j) : (2 * j + 8)) + 8 * half;
            b.u[j] = *(const unsigned int*)(brow + kb);
        }
        c = __builtin_amdgcn_wmma_f32_16x16x32_bf16(
                /*neg_a=*/false, a.v, /*neg_b=*/false, b.v,
                /*c_mod=*/(short)0, c, /*reuse_a=*/false, /*reuse_b=*/false);
    }

    #pragma unroll
    for (int r = 0; r < 8; ++r) obase[(r + 8 * half) * NN] = c[r];
}

// ---------------- Launcher ----------------
extern "C" void kernel_launch(void* const* d_in, const int* in_sizes, int n_in,
                              void* d_out, int out_size, void* d_ws, size_t ws_size,
                              hipStream_t stream) {
    const float* pred_logits = (const float*)d_in[0];   // (T*Q, C)
    const float* pred_boxes  = (const float*)d_in[1];   // (T*Q, 4)
    const int*   tgt_labels  = (const int*)d_in[2];     // (N*T,)
    const float* tgt_boxes   = (const float*)d_in[3];   // (N*T, 4)
    const int*   tgt_valid   = (const int*)d_in[4];     // (N*T,)
    const int*   tgt_ovalid  = (const int*)d_in[5];     // (N*T,)
    float* out = (float*)d_out;                         // (Q, N)

    // Workspace layout
    unsigned short* A_ws  = (unsigned short*)d_ws;                     // QQ*KPAD bf16 = 6.29 MB
    unsigned short* Bt_ws = A_ws + (size_t)QQ * KPAD;                  // NN*KPAD bf16 = 0.39 MB

    // 1) zero Bt + A pad columns
    init_ws_kernel<<<(NN * KPAD + 255) / 256, 256, 0, stream>>>(A_ws, Bt_ws);

    // 2) softmax -> bf16 A (one wave per row)
    softmax_kernel<<<(TT * QQ) / 8, 256, 0, stream>>>(pred_logits, A_ws);

    // 3) scatter one-hot class weights into Bt
    scatter_b_kernel<<<(NN * TT + 255) / 256, 256, 0, stream>>>(tgt_labels, tgt_valid,
                                                                tgt_ovalid, Bt_ws);

    // 4) bbox + iou costs -> d_out
    boxcost_kernel<<<(QQ * NN) / 256, 256, 0, stream>>>(pred_boxes, tgt_boxes,
                                                        tgt_valid, out);

    // 5) class cost GEMM accumulated on top of d_out (WMMA bf16)
    class_gemm_kernel<<<(QQ / 16) * (NN / 16) / 8, 256, 0, stream>>>(A_ws, Bt_ws, out);
}
